// GCPNet_16767552324117
// MI455X (gfx1250) — matrix operand order repaired and checked by reading, not compiled, not run on hardware
//
#include <hip/hip_runtime.h>
#include <hip/hip_bf16.h>
#include <math.h>

// ---------------- problem constants (from reference) ----------------
#define EMB   128
#define NV    4096
#define NC    2048
#define TMAX  16
#define GATE  512   // 4*EMB

typedef __bf16 bf16;
typedef __attribute__((ext_vector_type(8)))  bf16  v8bf;
typedef __attribute__((ext_vector_type(16))) bf16  v16bf;
typedef __attribute__((ext_vector_type(8)))  float v8f;
typedef int v4i __attribute__((vector_size(4 * sizeof(int))));  // matches builtin param

#define AS_GLOBAL __attribute__((address_space(1)))
#define AS_LOCAL  __attribute__((address_space(3)))

// Use CDNA5 async global->LDS path when the toolchain exposes it.
#if defined(__has_builtin)
#  if __has_builtin(__builtin_amdgcn_global_load_async_to_lds_b128) && \
      __has_builtin(__builtin_amdgcn_s_wait_asynccnt)
#    define USE_ASYNC_LDS 1
#  endif
#endif
#ifndef USE_ASYNC_LDS
#  define USE_ASYNC_LDS 0
#endif

// d_in index map: setup_inputs() insertion order, params dict flattened in
// _init_params insertion order.
enum {
  IN_M_VV = 0, IN_M_VC,
  IN_MLPC_W1, IN_MLPC_B1, IN_MLPC_W2, IN_MLPC_B2,
  IN_MLPV_W1, IN_MLPV_B1, IN_MLPV_W2, IN_MLPV_B2,
  IN_VOTE_W1, IN_VOTE_B1, IN_VOTE_W2, IN_VOTE_B2,
  IN_LV_WIH, IN_LV_WHH, IN_LV_GIH, IN_LV_BIH, IN_LV_GHH, IN_LV_BHH, IN_LV_GC, IN_LV_BC,
  IN_LC_WIH, IN_LC_WHH, IN_LC_GIH, IN_LC_BIH, IN_LC_GHH, IN_LC_BHH, IN_LC_GC, IN_LC_BC,
  IN_V_NORMAL, IN_C_RAND, IN_SLICE, IN_CN
};

__device__ __forceinline__ float sigm_f(float x){ return 1.0f/(1.0f+expf(-x)); }

// =====================================================================
// WMMA GEMM:  C[M,N] = A[M,K](bf16,row-major) * BT[N,K](bf16,row-major)^T
//
// Block = 256 threads = 8 waves, computes a 128x128 C tile (wave w owns the
// 16-row strip mT = mB*8+w; all 8 waves share one 128-column group nG).
// B panel is staged through LDS (double-buffered, async when available) so
// each B element is read from L2 once per block instead of once per wave.
// A fragments are register double-buffered; K loop is 2x unrolled
// (K must be a multiple of 64 -- true for 128/256/2048/4096).
// B fragments are software-pipelined across the 8 column tiles (load t+1
// into a fresh register before WMMA t) to overlap DS latency with WMMA.
// Fused epilogue: +bias (per column), ReLU, f32 out, bf16 row-major out,
// bf16 transposed out ([N,M]).
// =====================================================================
#define BROW 40   // LDS row stride in bf16 elements (80B, bank-spread)

__global__ void __launch_bounds__(256)
__attribute__((amdgpu_waves_per_eu(2)))
gemm_tn_wmma(const bf16* __restrict__ A, int lda,
             const bf16* __restrict__ BT, int ldb,
             int M, int N, int K,
             const float* __restrict__ bias, int relu,
             float* __restrict__ outF, int ldf,
             bf16*  __restrict__ outRM, int ldrm,
             bf16*  __restrict__ outTR, int ldtr)
{
  __shared__ __align__(16) bf16 ldsB[2][128 * BROW];

  const int mBlocks = M >> 7;                  // 128 rows per block
  const int mB   = blockIdx.x % mBlocks;
  const int nG   = blockIdx.x / mBlocks;
  const int wave = threadIdx.x >> 5;
  const int lane = threadIdx.x & 31;
  const int half = lane >> 4;                  // 0 or 1
  const int l16  = lane & 15;
  const int mT   = mB * 8 + wave;
  const int colBase = nG * 128;

  // A fragment source: lane holds row (mT*16+l16), K={half*8..+7, 16+half*8..+7}
  const bf16* aRow = A + (size_t)(mT * 16 + l16) * lda + half * 8;

  // ---- stage one 128x32 B panel into LDS buffer `which` (512 x 16B chunks)
  auto stage = [&](int k0, int which) {
    bf16* buf = &ldsB[which][0];
    #pragma unroll
    for (int j = 0; j < 2; ++j) {
      int c    = (threadIdx.x << 1) | j;       // 0..511
      int col  = c >> 2;                       // 0..127
      int part = c & 3;                        // 4 x 8 bf16 per column
      const bf16* src = BT + (size_t)(colBase + col) * ldb + k0 + part * 8;
      bf16* dst = buf + col * BROW + part * 8;
#if USE_ASYNC_LDS
      __builtin_amdgcn_global_load_async_to_lds_b128(
          (AS_GLOBAL v4i*)src, (AS_LOCAL v4i*)dst, 0, 0);
#else
      *(uint4*)dst = *(const uint4*)src;
#endif
    }
  };
  auto await_stage = [&]() {
#if USE_ASYNC_LDS
    __builtin_amdgcn_s_wait_asynccnt(0);
#endif
    __syncthreads();
  };
  auto loadA = [&](int k0) -> v16bf {
    v8bf a0 = *(const v8bf*)(aRow + k0);
    v8bf a1 = *(const v8bf*)(aRow + k0 + 16);
    v16bf f;
    #pragma unroll
    for (int i = 0; i < 8; ++i) { f[i] = a0[i]; f[8 + i] = a1[i]; }
    return f;
  };
  auto loadB = [&](const bf16* buf, int t) -> v16bf {
    const bf16* bp = buf + (t * 16 + l16) * BROW + half * 8;
    v8bf b0 = *(const v8bf*)bp;
    v8bf b1 = *(const v8bf*)(bp + 16);
    v16bf f;
    #pragma unroll
    for (int i = 0; i < 8; ++i) { f[i] = b0[i]; f[8 + i] = b1[i]; }
    return f;
  };

  v8f acc[8];
  #pragma unroll
  for (int t = 0; t < 8; ++t) {
    float bv = bias ? bias[colBase + t * 16 + l16] : 0.0f;
    #pragma unroll
    for (int r = 0; r < 8; ++r) acc[t][r] = bv;
  }

  // software-pipelined 8-tile WMMA chain: load fragment t+1 before WMMA t
  auto wmma8 = [&](v16bf fa, const bf16* buf) {
    v16bf cur = loadB(buf, 0);
    #pragma unroll
    for (int t = 0; t < 8; ++t) {
      v16bf nxt;
      if (t < 7) nxt = loadB(buf, t + 1);
      acc[t] = __builtin_amdgcn_wmma_f32_16x16x32_bf16(
          false, fa, false, cur, (short)0, acc[t], false, false);
      cur = nxt;
    }
  };

  // ---- pipelined main loop: stage k+32 / prefetch A while computing k ----
  stage(0, 0);
  v16bf faC = loadA(0);
  v16bf faN;
  for (int k0 = 0; k0 < K; k0 += 64) {
    await_stage();                 // buf0 ready; buf1 reads from prev iter done
    stage(k0 + 32, 1);             // k0+32 < K always (K % 64 == 0)
    faN = loadA(k0 + 32);
    wmma8(faC, &ldsB[0][0]);
    await_stage();                 // buf1 ready; buf0 reads above done
    if (k0 + 64 < K) {
      stage(k0 + 64, 0);
      faC = loadA(k0 + 64);
    }
    wmma8(faN, &ldsB[1][0]);
  }

  // C/D layout: vgpr r -> row mT*16 + half*8 + r, col = l16-based
  int row0 = mT * 16 + half * 8;
  #pragma unroll
  for (int t = 0; t < 8; ++t) {
    int col = colBase + t * 16 + l16;
    #pragma unroll
    for (int r = 0; r < 8; ++r) {
      float v = acc[t][r];
      if (relu) v = v > 0.0f ? v : 0.0f;
      int row = row0 + r;
      if (outF)  outF [(size_t)row * ldf  + col] = v;
      if (outRM) outRM[(size_t)row * ldrm + col] = (bf16)v;
      if (outTR) outTR[(size_t)col * ldtr + row] = (bf16)v;
    }
  }
}

// =====================================================================
// Fused LayerNorm-LSTM cell (one block of 256 threads per row).
//   g = LN(gih; g_ih,b_ih) + LN(ghh; g_hh,b_hh)            (width 512)
//   i,f,gg,o = split(g); c2 = sig(f)*c + sig(i)*tanh(gg)
//   h2 = sig(o) * tanh(LN(c2; g_c,b_c))                    (width 128)
// Writes c (f32), h (f32 -> d_out region), h bf16 row-major, h bf16 ^T.
// =====================================================================
__device__ __forceinline__ float blockSum256(float v, float* red) {
  int t = threadIdx.x;
  red[t] = v; __syncthreads();
  #pragma unroll
  for (int off = 128; off > 0; off >>= 1) {
    if (t < off) red[t] += red[t + off];
    __syncthreads();
  }
  float r = red[0]; __syncthreads();
  return r;
}

__global__ void __launch_bounds__(256)
lnlstm_kernel(const float* __restrict__ GIH, const float* __restrict__ GHH,
              const float* __restrict__ g_ih, const float* __restrict__ b_ih,
              const float* __restrict__ g_hh, const float* __restrict__ b_hh,
              const float* __restrict__ g_c,  const float* __restrict__ b_c,
              float* __restrict__ cSt, float* __restrict__ hOut,
              bf16* __restrict__ hRM, bf16* __restrict__ hTR, int ldtr)
{
  __shared__ float red[256];
  __shared__ float gbuf[GATE];
  int row = blockIdx.x, t = threadIdx.x;
  const float* gih = GIH + (size_t)row * GATE;
  const float* ghh = GHH + (size_t)row * GATE;
  float a0 = gih[t], a1 = gih[t + 256];
  float h0 = ghh[t], h1 = ghh[t + 256];

  float sum_i = blockSum256(a0 + a1, red);
  float ssq_i = blockSum256(a0 * a0 + a1 * a1, red);
  float sum_h = blockSum256(h0 + h1, red);
  float ssq_h = blockSum256(h0 * h0 + h1 * h1, red);
  const float inv512 = 1.0f / 512.0f;
  float mi = sum_i * inv512, vi = ssq_i * inv512 - mi * mi;
  float mh = sum_h * inv512, vh = ssq_h * inv512 - mh * mh;
  float ri = rsqrtf(vi + 1e-5f), rh = rsqrtf(vh + 1e-5f);

  gbuf[t]       = (a0 - mi) * ri * g_ih[t]       + b_ih[t]
                + (h0 - mh) * rh * g_hh[t]       + b_hh[t];
  gbuf[t + 256] = (a1 - mi) * ri * g_ih[t + 256] + b_ih[t + 256]
                + (h1 - mh) * rh * g_hh[t + 256] + b_hh[t + 256];
  __syncthreads();

  float c2 = 0.0f;
  if (t < EMB) {
    float gi = gbuf[t], gf = gbuf[EMB + t], gg = gbuf[2 * EMB + t];
    c2 = sigm_f(gf) * cSt[(size_t)row * EMB + t] + sigm_f(gi) * tanhf(gg);
    cSt[(size_t)row * EMB + t] = c2;
  }
  float sc = blockSum256(t < EMB ? c2 : 0.0f, red);
  float sq = blockSum256(t < EMB ? c2 * c2 : 0.0f, red);
  const float inv128 = 1.0f / 128.0f;
  float mc = sc * inv128, vc = sq * inv128 - mc * mc;
  float rc = rsqrtf(vc + 1e-5f);
  if (t < EMB) {
    float go = gbuf[3 * EMB + t];
    float ln = (c2 - mc) * rc * g_c[t] + b_c[t];
    float h2 = sigm_f(go) * tanhf(ln);
    hOut[(size_t)row * EMB + t] = h2;
    hRM [(size_t)row * EMB + t] = (bf16)h2;
    hTR [(size_t)t * ldtr + row] = (bf16)h2;
  }
}

// =====================================================================
// setup / pointwise helpers
// =====================================================================
__global__ void cvt_bf16(const float* __restrict__ s, bf16* __restrict__ d, int n) {
  int i = blockIdx.x * blockDim.x + threadIdx.x;
  if (i < n) d[i] = (bf16)s[i];
}

__global__ void cvt_bf16_t(const float* __restrict__ s, bf16* __restrict__ d,
                           int rows, int cols) {   // d[c*rows+r] = s[r*cols+c]
  int i = blockIdx.x * blockDim.x + threadIdx.x;
  if (i >= rows * cols) return;
  int r = i / cols, c = i - r * cols;
  d[(size_t)c * rows + r] = (bf16)s[i];
}

__global__ void init_state(const float* __restrict__ initv, int n,
                           bf16* __restrict__ hRM, bf16* __restrict__ hTR,
                           float* __restrict__ cSt, float* __restrict__ hOut) {
  int idx = blockIdx.x * blockDim.x + threadIdx.x;
  if (idx >= n * EMB) return;
  int i = idx / EMB, j = idx - i * EMB;
  float v = initv[j];
  hRM[idx] = (bf16)v;
  hTR[(size_t)j * n + i] = (bf16)v;
  cSt[idx] = 0.0f;
  hOut[idx] = v;
}

__global__ void vote_score(const bf16* __restrict__ H1,
                           const float* __restrict__ W2, const float* __restrict__ b2,
                           float* __restrict__ scores) {
  int i = blockIdx.x * blockDim.x + threadIdx.x;
  if (i >= NV) return;
  const bf16* r = H1 + (size_t)i * EMB;
  float s = 0.0f;
  #pragma unroll 8
  for (int j = 0; j < EMB; ++j) s += (float)r[j] * W2[j];
  scores[i] = s + b2[0];
}

__global__ void finalize_kernel(const float* __restrict__ scores, int ng,
                                float* __restrict__ out) {
  int slice = NV / ng;
  for (int g = threadIdx.x; g < ng; g += blockDim.x) {
    float s = 0.0f;
    for (int k = 0; k < slice; ++k) s += scores[g * slice + k];
    float m = s / (float)slice;
    out[g]      = 1.0f / (1.0f + expf(-m));  // pred
    out[ng + g] = m;                         // means
  }
}

// =====================================================================
// host side
// =====================================================================
extern "C" void kernel_launch(void* const* d_in, const int* in_sizes, int n_in,
                              void* d_out, int out_size, void* d_ws, size_t ws_size,
                              hipStream_t stream) {
  (void)in_sizes; (void)n_in; (void)ws_size;
  const float* Mvv   = (const float*)d_in[IN_M_VV];
  const float* Mvc   = (const float*)d_in[IN_M_VC];
  float* outF = (float*)d_out;

  // n_graphs derived from out_size: out = pred[ng] ++ means[ng] ++ V ++ C
  int ng = (out_size - (NV + NC) * EMB) / 2;
  if (ng < 1) ng = 1;
  float* hOutV = outF + 2 * ng;               // final V lives in d_out
  float* hOutC = hOutV + (size_t)NV * EMB;    // final C lives in d_out

  // ---- workspace carving (deterministic) ----
  char* ws = (char*)d_ws;
  size_t off = 0;
  auto carve = [&](size_t bytes) -> void* {
    off = (off + 255) & ~(size_t)255;
    void* p = ws + off;
    off += bytes;
    return p;
  };
  bf16* Mvv_bf  = (bf16*)carve((size_t)NV * NV * 2);
  bf16* Mvc_bf  = (bf16*)carve((size_t)NV * NC * 2);
  bf16* Mcv_bf  = (bf16*)carve((size_t)NV * NC * 2);   // M_vc^T, [NC,NV]
  bf16* W1v_bf  = (bf16*)carve(EMB * EMB * 2);
  bf16* W2v_bf  = (bf16*)carve(EMB * EMB * 2);
  bf16* W1c_bf  = (bf16*)carve(EMB * EMB * 2);
  bf16* W2c_bf  = (bf16*)carve(EMB * EMB * 2);
  bf16* Wvote_bf= (bf16*)carve(EMB * EMB * 2);
  bf16* WihV_bf = (bf16*)carve(GATE * 2 * EMB * 2);
  bf16* WhhV_bf = (bf16*)carve(GATE * EMB * 2);
  bf16* WihC_bf = (bf16*)carve(GATE * EMB * 2);
  bf16* WhhC_bf = (bf16*)carve(GATE * EMB * 2);
  bf16* hV_rm   = (bf16*)carve((size_t)NV * EMB * 2);
  bf16* hV_tr   = (bf16*)carve((size_t)NV * EMB * 2);
  bf16* hC_rm   = (bf16*)carve((size_t)NC * EMB * 2);
  bf16* hC_tr   = (bf16*)carve((size_t)NC * EMB * 2);
  bf16* H1V_bf  = (bf16*)carve((size_t)NV * EMB * 2);
  bf16* H1C_bf  = (bf16*)carve((size_t)NC * EMB * 2);
  bf16* Vm_tr   = (bf16*)carve((size_t)NV * EMB * 2);  // [EMB,NV]
  bf16* Cm_tr   = (bf16*)carve((size_t)NC * EMB * 2);  // [EMB,NC]
  bf16* xv_bf   = (bf16*)carve((size_t)NV * 2 * EMB * 2); // [NV,256]
  bf16* XC_bf   = (bf16*)carve((size_t)NC * EMB * 2);
  float* GIHV   = (float*)carve((size_t)NV * GATE * 4);
  float* GHHV   = (float*)carve((size_t)NV * GATE * 4);
  float* GIHC   = (float*)carve((size_t)NC * GATE * 4);
  float* GHHC   = (float*)carve((size_t)NC * GATE * 4);
  float* cV     = (float*)carve((size_t)NV * EMB * 4);
  float* cC     = (float*)carve((size_t)NC * EMB * 4);
  float* scores = (float*)carve((size_t)NV * 4);

  auto cvt = [&](const void* src, bf16* dst, int n) {
    cvt_bf16<<<(n + 255) / 256, 256, 0, stream>>>((const float*)src, dst, n);
  };
  // one-time-per-call bf16 conversions (deterministic, graph-capture safe)
  cvt(Mvv, Mvv_bf, NV * NV);
  cvt(Mvc, Mvc_bf, NV * NC);
  cvt_bf16_t<<<(NV * NC + 255) / 256, 256, 0, stream>>>(Mvc, Mcv_bf, NV, NC);
  cvt(d_in[IN_MLPV_W1], W1v_bf, EMB * EMB);
  cvt(d_in[IN_MLPV_W2], W2v_bf, EMB * EMB);
  cvt(d_in[IN_MLPC_W1], W1c_bf, EMB * EMB);
  cvt(d_in[IN_MLPC_W2], W2c_bf, EMB * EMB);
  cvt(d_in[IN_VOTE_W1], Wvote_bf, EMB * EMB);
  cvt(d_in[IN_LV_WIH], WihV_bf, GATE * 2 * EMB);
  cvt(d_in[IN_LV_WHH], WhhV_bf, GATE * EMB);
  cvt(d_in[IN_LC_WIH], WihC_bf, GATE * EMB);
  cvt(d_in[IN_LC_WHH], WhhC_bf, GATE * EMB);

  init_state<<<(NV * EMB + 255) / 256, 256, 0, stream>>>(
      (const float*)d_in[IN_V_NORMAL], NV, hV_rm, hV_tr, cV, hOutV);
  init_state<<<(NC * EMB + 255) / 256, 256, 0, stream>>>(
      (const float*)d_in[IN_C_RAND], NC, hC_rm, hC_tr, cC, hOutC);

  auto gemm = [&](const bf16* A, int lda, const bf16* BT, int ldb,
                  int M, int N, int K, const float* bias, int relu,
                  float* oF, int ldf, bf16* oRM, int ldrm, bf16* oTR, int ldtr) {
    int blocks = (M / 128) * (N / 128);
    gemm_tn_wmma<<<blocks, 256, 0, stream>>>(
        A, lda, BT, ldb, M, N, K, bias, relu, oF, ldf, oRM, ldrm, oTR, ldtr);
  };

  const float* b1v = (const float*)d_in[IN_MLPV_B1];
  const float* b2v = (const float*)d_in[IN_MLPV_B2];
  const float* b1c = (const float*)d_in[IN_MLPC_B1];
  const float* b2c = (const float*)d_in[IN_MLPC_B2];

  for (int step = 0; step < TMAX; ++step) {
    // Vm = MLP(hV); emit Vm^T bf16 for xc GEMM
    gemm(hV_rm, EMB, W1v_bf, EMB, NV, EMB, EMB, b1v, 1,
         nullptr, 0, H1V_bf, EMB, nullptr, 0);
    gemm(H1V_bf, EMB, W2v_bf, EMB, NV, EMB, EMB, b2v, 0,
         nullptr, 0, nullptr, 0, Vm_tr, NV);
    // Cm = MLP(hC); emit Cm^T bf16 for xv GEMM
    gemm(hC_rm, EMB, W1c_bf, EMB, NC, EMB, EMB, b1c, 1,
         nullptr, 0, H1C_bf, EMB, nullptr, 0);
    gemm(H1C_bf, EMB, W2c_bf, EMB, NC, EMB, EMB, b2c, 0,
         nullptr, 0, nullptr, 0, Cm_tr, NC);
    // xv = [ M_vv @ hV , M_vc @ Cm ]  -> bf16 [NV,256], fused concat via ld
    gemm(Mvv_bf, NV, hV_tr, NV, NV, EMB, NV, nullptr, 0,
         nullptr, 0, xv_bf, 2 * EMB, nullptr, 0);
    gemm(Mvc_bf, NC, Cm_tr, NC, NV, EMB, NC, nullptr, 0,
         nullptr, 0, xv_bf + EMB, 2 * EMB, nullptr, 0);
    // V-LSTM gate pre-activations
    gemm(xv_bf, 2 * EMB, WihV_bf, 2 * EMB, NV, GATE, 2 * EMB, nullptr, 0,
         GIHV, GATE, nullptr, 0, nullptr, 0);
    gemm(hV_rm, EMB, WhhV_bf, EMB, NV, GATE, EMB, nullptr, 0,
         GHHV, GATE, nullptr, 0, nullptr, 0);
    lnlstm_kernel<<<NV, 256, 0, stream>>>(
        GIHV, GHHV,
        (const float*)d_in[IN_LV_GIH], (const float*)d_in[IN_LV_BIH],
        (const float*)d_in[IN_LV_GHH], (const float*)d_in[IN_LV_BHH],
        (const float*)d_in[IN_LV_GC],  (const float*)d_in[IN_LV_BC],
        cV, hOutV, hV_rm, hV_tr, NV);
    // xc = M_vc^T @ Vm   (uses old-hV-derived Vm, correct ordering)
    gemm(Mcv_bf, NV, Vm_tr, NV, NC, EMB, NV, nullptr, 0,
         nullptr, 0, XC_bf, EMB, nullptr, 0);
    // C-LSTM gate pre-activations
    gemm(XC_bf, EMB, WihC_bf, EMB, NC, GATE, EMB, nullptr, 0,
         GIHC, GATE, nullptr, 0, nullptr, 0);
    gemm(hC_rm, EMB, WhhC_bf, EMB, NC, GATE, EMB, nullptr, 0,
         GHHC, GATE, nullptr, 0, nullptr, 0);
    lnlstm_kernel<<<NC, 256, 0, stream>>>(
        GIHC, GHHC,
        (const float*)d_in[IN_LC_GIH], (const float*)d_in[IN_LC_BIH],
        (const float*)d_in[IN_LC_GHH], (const float*)d_in[IN_LC_BHH],
        (const float*)d_in[IN_LC_GC],  (const float*)d_in[IN_LC_BC],
        cC, hOutC, hC_rm, hC_tr, NC);
  }

  // vote: hidden = relu(hV @ voteW1^T + b1) -> per-vertex score -> per-graph
  gemm(hV_rm, EMB, Wvote_bf, EMB, NV, EMB, EMB,
       (const float*)d_in[IN_VOTE_B1], 1, nullptr, 0, H1V_bf, EMB, nullptr, 0);
  vote_score<<<(NV + 255) / 256, 256, 0, stream>>>(
      H1V_bf, (const float*)d_in[IN_VOTE_W2], (const float*)d_in[IN_VOTE_B2], scores);
  finalize_kernel<<<1, 256, 0, stream>>>(scores, ng, outF);
}
